// GoSpecificWattiPooling_44032004719131
// MI455X (gfx1250) — matrix operand order repaired and checked by reading, not compiled, not run on hardware
//
#include <hip/hip_runtime.h>
#include <hip/hip_bf16.h>

// ---------------------------------------------------------------------------
// Problem constants (from reference): B=8, L=4096, T=128, D_H=1024, D_G=256,
// D_PROJ=256.  scale = D_PROJ^-0.5 = 1/16 (folded into Q projection).
// ---------------------------------------------------------------------------
constexpr int Bn = 8;
constexpr int Ln = 4096;
constexpr int Tn = 128;
constexpr int DH = 1024;
constexpr int DG = 256;
constexpr int DP = 256;

typedef unsigned short u16;
typedef __bf16 v16bf __attribute__((ext_vector_type(16)));
typedef __bf16 v8bf  __attribute__((ext_vector_type(8)));
typedef float  v8f   __attribute__((ext_vector_type(8)));

// Workspace layout (bytes). Everything 16B aligned.
constexpr size_t N_H    = (size_t)Bn * Ln * DH;          // 33,554,432 elems
constexpr size_t N_WK   = (size_t)DH * DP;               // 262,144
constexpr size_t OFF_HB  = 0;                            // bf16 H      (67.1 MB)
constexpr size_t OFF_KT  = OFF_HB  + N_H * 2;            // bf16 K^T    (16.8 MB)  [b][p][l]
constexpr size_t OFF_LG  = OFF_KT  + (size_t)Bn * DP * Ln * 2;   // f32 logits (16.8 MB) [b][t][l]
constexpr size_t OFF_QB  = OFF_LG  + (size_t)Bn * Tn * Ln * 4;   // bf16 Q (0.5 MB) [b][t][p]
constexpr size_t OFF_WKB = OFF_QB  + (size_t)Bn * Tn * DP * 2;   // bf16 Wk (0.5 MB)
constexpr size_t OFF_ST  = OFF_WKB + N_WK * 2;                   // f32 stats [B*T][2]

// ---------------------------------------------------------------------------
// CDNA5 async global->LDS copy (ASYNCcnt path), with compile-safe fallback.
// The builtin's parameters are pointers to `int __vector(4)` in AS(1)/AS(3)
// (exact pointee type taken from the clang diagnostic in the previous round).
// ---------------------------------------------------------------------------
#if defined(__gfx1250__) && __has_builtin(__builtin_amdgcn_global_load_async_to_lds_b128)
#define HAVE_ASYNC_LDS 1
#else
#define HAVE_ASYNC_LDS 0
#endif

typedef int v4i_vs __attribute__((vector_size(16)));
typedef __attribute__((address_space(1))) v4i_vs* gptr4;   // global int4*
typedef __attribute__((address_space(3))) v4i_vs* lptr4;   // LDS int4*

__device__ __forceinline__ void async_copy16(const u16* g, u16* l) {
#if HAVE_ASYNC_LDS
  __builtin_amdgcn_global_load_async_to_lds_b128(
      (gptr4)g, (lptr4)l, /*offset=*/0, /*cpol=*/0);
#else
  *reinterpret_cast<uint4*>(l) = *reinterpret_cast<const uint4*>(g);
#endif
}

__device__ __forceinline__ void stage_sync() {
#if HAVE_ASYNC_LDS
#if __has_builtin(__builtin_amdgcn_s_wait_asynccnt)
  __builtin_amdgcn_s_wait_asynccnt(0);
#else
  asm volatile("s_wait_asynccnt 0" ::: "memory");
#endif
#endif
  __syncthreads();
}

// ---------------------------------------------------------------------------
// Helpers
// ---------------------------------------------------------------------------
__device__ __forceinline__ u16 bf16bits(float f) {
  return __builtin_bit_cast(u16, static_cast<__bf16>(f));
}

// Build a 16-element bf16 fragment from two contiguous 16B chunks.
// (A-matrix: chunk0 = K 8h+0..7 -> VGPR0..3, chunk1 = K 16+8h+0..7 -> VGPR4..7.
//  B-matrix (load_frag_col): same K order, one N column per lane.)
__device__ __forceinline__ v16bf mk_frag(uint4 lo, uint4 hi) {
  v8bf l = __builtin_bit_cast(v8bf, lo);
  v8bf h = __builtin_bit_cast(v8bf, hi);
  return __builtin_shufflevector(l, h, 0, 1, 2, 3, 4, 5, 6, 7,
                                       8, 9, 10, 11, 12, 13, 14, 15);
}

__device__ __forceinline__ v16bf load_frag_g(const u16* p0, const u16* p1) {
  return mk_frag(*reinterpret_cast<const uint4*>(p0),
                 *reinterpret_cast<const uint4*>(p1));
}

__device__ __forceinline__ v16bf load_frag_col(const u16* p, size_t pitch) {
  v16bf r;
#pragma unroll
  for (int j = 0; j < 8; ++j) {
    r[j]     = __builtin_bit_cast(__bf16, p[(size_t)j * pitch]);
    r[8 + j] = __builtin_bit_cast(__bf16, p[(size_t)(16 + j) * pitch]);
  }
  return r;
}

__device__ __forceinline__ v8f wmma_bf16(v16bf a, v16bf b, v8f c) {
  // D = A(16x32) * B(32x16) + C, f32 accumulate.
  return __builtin_amdgcn_wmma_f32_16x16x32_bf16(
      false, a, false, b, (short)0, c, false, false);
}

// ---------------------------------------------------------------------------
// Kernel 1/2: f32 -> bf16 pack (8 elems/thread, 16B stores)
// ---------------------------------------------------------------------------
__global__ __launch_bounds__(256) void k_pack_bf16(const float* __restrict__ src,
                                                   u16* __restrict__ dst) {
  const size_t i = (size_t)blockIdx.x * 256 + threadIdx.x;
  const float4 f0 = reinterpret_cast<const float4*>(src)[i * 2 + 0];
  const float4 f1 = reinterpret_cast<const float4*>(src)[i * 2 + 1];
  v8bf o;
  o[0] = (__bf16)f0.x; o[1] = (__bf16)f0.y; o[2] = (__bf16)f0.z; o[3] = (__bf16)f0.w;
  o[4] = (__bf16)f1.x; o[5] = (__bf16)f1.y; o[6] = (__bf16)f1.z; o[7] = (__bf16)f1.w;
  reinterpret_cast<uint4*>(dst)[i] = __builtin_bit_cast(uint4, o);
}

// ---------------------------------------------------------------------------
// Kernel 3: Q = G * Wq * scale  -> bf16.  Tiny (0.13 GFLOP): scalar is fine.
// ---------------------------------------------------------------------------
__global__ __launch_bounds__(256) void k_qproj(const float* __restrict__ G,
                                               const float* __restrict__ Wq,
                                               u16* __restrict__ Qb) {
  const int idx = blockIdx.x * 256 + threadIdx.x;   // 0 .. B*T*DP-1
  const int row = idx >> 8;                          // b*T + t
  const int p   = idx & (DP - 1);
  const float* g = G + (size_t)row * DG;
  float s = 0.f;
#pragma unroll 8
  for (int k = 0; k < DG; ++k) s = fmaf(g[k], Wq[(size_t)k * DP + p], s);
  Qb[idx] = bf16bits(s * 0.0625f);                   // scale = DP^-0.5
}

// ---------------------------------------------------------------------------
// Kernel 4: K^T projection.  KT[b][p][l] (bf16) = (Hb[b*L+l][:] . Wkb[:][p]).
// M = B*L = 32768, K = 1024, N = 256.  WG = 256 thr = 8 waves (4M x 2N),
// WG tile 128x64, wave tile 32x32 (2x2 WMMA tiles). Grid (256, 4).
// Double-buffered async global->LDS staging, 64-deep K stages:
//   stage copy (ASYNCcnt) of stage s+1 overlaps the 8 WMMAs of stage s.
// LDS pitch 72 (144B = 9*16B) keeps 16B alignment and breaks bank patterns.
// ---------------------------------------------------------------------------
constexpr int APITCH = 72;
constexpr int BPITCH = 72;

__global__ __launch_bounds__(256) void k_proj_kt(const u16* __restrict__ Hb,
                                                 const u16* __restrict__ Wkb,
                                                 u16* __restrict__ KT) {
  __shared__ u16 __attribute__((aligned(16))) sA[2][128 * APITCH];  // 36 KB
  __shared__ u16 __attribute__((aligned(16))) sB[2][64 * BPITCH];   // 18 KB

  const int tid  = threadIdx.x;
  const int lane = tid & 31;
  const int wave = tid >> 5;
  const int wm = wave & 3, wn = wave >> 2;
  const int Mb0 = blockIdx.x * 128;
  const int Nb0 = blockIdx.y * 64;
  const int mrow = lane & 15;          // A-row / D-col index
  const int kh   = lane >> 4;          // K-half select

  // Issue async copies for k-range [ks, ks+64) into stage buffer s.
  auto issue = [&](int ks, int s) {
    // A tile: 128 rows x 64 K, 1024 x 16B chunks, 4 per thread.
#pragma unroll
    for (int j = 0; j < 4; ++j) {
      const int c = tid * 4 + j;
      const int row = c >> 3, cc = (c & 7) * 8;
      async_copy16(Hb + (size_t)(Mb0 + row) * DH + ks + cc,
                   &sA[s][row * APITCH + cc]);
    }
    // B tile: 64 K-rows x 64 N, 512 x 16B chunks, 2 per thread.
#pragma unroll
    for (int j = 0; j < 2; ++j) {
      const int c = tid * 2 + j;
      const int row = c >> 3, cc = (c & 7) * 8;
      async_copy16(Wkb + (size_t)(ks + row) * DP + Nb0 + cc,
                   &sB[s][row * BPITCH + cc]);
    }
  };

  v8f acc[2][2] = {};
  issue(0, 0);
  for (int ks = 0; ks < DH; ks += 64) {
    const int s = (ks >> 6) & 1;
    stage_sync();                        // drain ASYNCcnt, publish stage s
    if (ks + 64 < DH) issue(ks + 64, s ^ 1);   // overlap copy with WMMAs

#pragma unroll
    for (int sub = 0; sub < 2; ++sub) {  // two 32-deep WMMA K-steps
      const int kk0 = sub * 32 + 8 * kh;
      v16bf a[2], bf[2];
#pragma unroll
      for (int mt = 0; mt < 2; ++mt) {
        const u16* ap = &sA[s][(wm * 32 + mt * 16 + mrow) * APITCH + kk0];
        a[mt] = mk_frag(*reinterpret_cast<const uint4*>(ap),
                        *reinterpret_cast<const uint4*>(ap + 16));
      }
#pragma unroll
      for (int nt = 0; nt < 2; ++nt) {
        const u16* bp = &sB[s][(sub * 32 + 8 * kh) * BPITCH + wn * 32 + nt * 16 + mrow];
        bf[nt] = load_frag_col(bp, BPITCH);
      }
#pragma unroll
      for (int mt = 0; mt < 2; ++mt)
#pragma unroll
        for (int nt = 0; nt < 2; ++nt)
          acc[mt][nt] = wmma_bf16(a[mt], bf[nt], acc[mt][nt]);
    }
  }

  // Transposed bf16 store: D(m,n) -> KT[b][p=n][l].  Consecutive v are
  // consecutive l, so the compiler merges these into b128 stores per lane.
#pragma unroll
  for (int mt = 0; mt < 2; ++mt)
#pragma unroll
    for (int nt = 0; nt < 2; ++nt) {
      const int p = Nb0 + wn * 32 + nt * 16 + mrow;
#pragma unroll
      for (int v = 0; v < 8; ++v) {
        const int m_g = Mb0 + wm * 32 + mt * 16 + v + 8 * kh;
        const int bb = m_g >> 12;            // / L
        const int l  = m_g & (Ln - 1);
        KT[(size_t)bb * DP * Ln + (size_t)p * Ln + l] = bf16bits(acc[mt][nt][v]);
      }
    }
}

// ---------------------------------------------------------------------------
// Kernel 5: logits[b][t][l] = Qb[b][t][:] . KT[b][:][l], mask -> -1e30.
// Per WG: 16(T) x 256(L) tile; wave w owns 32 L-cols (2 WMMA tiles).
// Grid (L/256 = 16, B*T/16 = 64).
// ---------------------------------------------------------------------------
__global__ __launch_bounds__(256) void k_logits(const u16* __restrict__ Qb,
                                                const u16* __restrict__ KT,
                                                const int* __restrict__ mask,
                                                float* __restrict__ logits) {
  const int lane = threadIdx.x & 31;
  const int wave = threadIdx.x >> 5;
  const int b  = blockIdx.y >> 3;
  const int t0 = (blockIdx.y & 7) * 16;
  const int lw = blockIdx.x * 256 + wave * 32;
  const int m  = lane & 15;
  const int kh = lane >> 4;

  v8f acc[2] = {};
  for (int ps = 0; ps < DP / 32; ++ps) {
    const int kk0 = ps * 32 + 8 * kh;
    const u16* ap = Qb + (size_t)(b * Tn + t0 + m) * DP + kk0;
    const v16bf a = load_frag_g(ap, ap + 16);
#pragma unroll
    for (int lt = 0; lt < 2; ++lt) {
      const u16* bp = KT + (size_t)b * DP * Ln + (size_t)(ps * 32 + 8 * kh) * Ln
                      + lw + lt * 16 + m;
      acc[lt] = wmma_bf16(a, load_frag_col(bp, Ln), acc[lt]);
    }
  }
#pragma unroll
  for (int lt = 0; lt < 2; ++lt) {
    const int lcol = lw + lt * 16 + m;
    const bool msk = mask[b * Ln + lcol] != 0;
#pragma unroll
    for (int v = 0; v < 8; ++v) {
      const float s = msk ? -1.0e30f : acc[lt][v];
      logits[(size_t)(b * Tn + t0 + v + 8 * kh) * Ln + lcol] = s;
    }
  }
}

// ---------------------------------------------------------------------------
// Kernel 6: per-row softmax stats: stats[r] = {rowmax, 1/sum(exp(x-max))}.
// One wave per row of 4096.  Grid 128 x 256thr.
// ---------------------------------------------------------------------------
__global__ __launch_bounds__(256) void k_stats(const float* __restrict__ logits,
                                               float* __restrict__ stats) {
  const int lane = threadIdx.x & 31;
  const int wave = threadIdx.x >> 5;
  const int row = blockIdx.x * 8 + wave;   // 0 .. B*T-1
  const float* p = logits + (size_t)row * Ln;
  float mx = -3.0e38f;
  for (int i = lane; i < Ln; i += 32) mx = fmaxf(mx, p[i]);
#pragma unroll
  for (int o = 16; o; o >>= 1) mx = fmaxf(mx, __shfl_xor(mx, o, 32));
  float s = 0.f;
  for (int i = lane; i < Ln; i += 32) s += __expf(p[i] - mx);
#pragma unroll
  for (int o = 16; o; o >>= 1) s += __shfl_xor(s, o, 32);
  if (lane == 0) { stats[row * 2] = mx; stats[row * 2 + 1] = 1.0f / s; }
}

// ---------------------------------------------------------------------------
// Kernel 7: Z[b][t][d] = sum_l alpha[b][t][l] * H[b][l][d]  (bf16 WMMA).
// Per WG: 16(T) x 128(D); wave w owns one 16-col D tile (8 accum VGPRs).
// alpha is computed once per WG per 256-L chunk and staged in LDS as bf16;
// A-fragments come from LDS (ds_load_b128), B-fragments from Hb (global),
// with global_prefetch of the next chunk's Hb rows.
// Grid (DH/128 = 8, B*T/16 = 64).
// ---------------------------------------------------------------------------
__global__ __launch_bounds__(256) void k_attn_out(const float* __restrict__ logits,
                                                  const float* __restrict__ stats,
                                                  const u16* __restrict__ Hb,
                                                  float* __restrict__ Z) {
  __shared__ float s_max[16];
  __shared__ float s_inv[16];
  __shared__ u16 __attribute__((aligned(16))) s_alpha[16 * 256];  // 8 KB

  const int lane = threadIdx.x & 31;
  const int wave = threadIdx.x >> 5;
  const int b  = blockIdx.y >> 3;
  const int t0 = (blockIdx.y & 7) * 16;
  const int d0 = blockIdx.x * 128 + wave * 16;
  const int m  = lane & 15;
  const int kh = lane >> 4;

  if (threadIdx.x < 16) {
    s_max[threadIdx.x] = stats[(b * Tn + t0 + threadIdx.x) * 2];
    s_inv[threadIdx.x] = stats[(b * Tn + t0 + threadIdx.x) * 2 + 1];
  }
  __syncthreads();

  v8f acc = {};
  for (int lc = 0; lc < Ln; lc += 256) {
    // Phase 1: materialize alpha (bf16) for this 16x256 chunk, once per WG.
    const int col = lc + wave * 32 + lane;
#pragma unroll
    for (int i = 0; i < 16; ++i) {
      const float v = logits[(size_t)(b * Tn + t0 + i) * Ln + col];
      const float al = __expf(v - s_max[i]) * s_inv[i];   // 0 for masked (-1e30)
      s_alpha[i * 256 + wave * 32 + lane] = bf16bits(al);
    }
    __syncthreads();

    // Phase 2: 8 WMMAs over the 256-L chunk; A from LDS, B from Hb.
    const bool pf = (lc + 256) < Ln;
#pragma unroll
    for (int ls = 0; ls < 8; ++ls) {
      const int kb = ls * 32 + 8 * kh;
      const u16* ap = s_alpha + m * 256 + kb;
      const v16bf a = mk_frag(*reinterpret_cast<const uint4*>(ap),
                              *reinterpret_cast<const uint4*>(ap + 16));
      const int lrow = lc + ls * 32 + 8 * kh;
      const u16* bp = Hb + (size_t)(b * Ln + lrow) * DH + d0 + m;
      if (pf) __builtin_prefetch(bp + 256 * DH, 0, 1);  // next chunk's row
      acc = wmma_bf16(a, load_frag_col(bp, DH), acc);
    }
    __syncthreads();
  }

#pragma unroll
  for (int v = 0; v < 8; ++v)
    Z[(size_t)(b * Tn + t0 + v + 8 * kh) * DH + d0 + m] = acc[v];
}

// ---------------------------------------------------------------------------
// Host launch
// ---------------------------------------------------------------------------
extern "C" void kernel_launch(void* const* d_in, const int* in_sizes, int n_in,
                              void* d_out, int out_size, void* d_ws, size_t ws_size,
                              hipStream_t stream) {
  (void)in_sizes; (void)n_in; (void)out_size; (void)ws_size;

  const float* H    = (const float*)d_in[0];   // [B, L, D_H]
  const float* G    = (const float*)d_in[1];   // [B, T, D_G]
  const int*   mask = (const int*)d_in[2];     // [B, L]
  const float* Wk   = (const float*)d_in[3];   // [D_H, D_PROJ]
  const float* Wq   = (const float*)d_in[4];   // [D_G, D_PROJ]
  float*       Z    = (float*)d_out;           // [B, T, D_H]

  char* ws = (char*)d_ws;
  u16*   Hb     = (u16*)(ws + OFF_HB);
  u16*   KT     = (u16*)(ws + OFF_KT);
  float* logits = (float*)(ws + OFF_LG);
  u16*   Qb     = (u16*)(ws + OFF_QB);
  u16*   Wkb    = (u16*)(ws + OFF_WKB);
  float* stats  = (float*)(ws + OFF_ST);

  // 1-2: pack H and Wk to bf16 (8 elems/thread)
  k_pack_bf16<<<(int)(N_H / 8 / 256), 256, 0, stream>>>(H, Hb);
  k_pack_bf16<<<(int)(N_WK / 8 / 256), 256, 0, stream>>>(Wk, Wkb);

  // 3: Q projection (scale folded in)
  k_qproj<<<(Bn * Tn * DP) / 256, 256, 0, stream>>>(G, Wq, Qb);

  // 4: K^T projection GEMM (WMMA bf16, async double-buffered LDS staging)
  k_proj_kt<<<dim3((Bn * Ln) / 128, DP / 64), 256, 0, stream>>>(Hb, Wkb, KT);

  // 5: masked logits GEMM (WMMA bf16)
  k_logits<<<dim3(Ln / 256, (Bn * Tn) / 16), 256, 0, stream>>>(Qb, KT, mask, logits);

  // 6: softmax row stats
  k_stats<<<(Bn * Tn) / 8, 256, 0, stream>>>(logits, stats);

  // 7: Z = alpha * H (WMMA bf16, alpha staged via LDS)
  k_attn_out<<<dim3(DH / 128, (Bn * Tn) / 16), 256, 0, stream>>>(logits, stats, Hb, Z);
}
